// NADE_52931176955982
// MI455X (gfx1250) — compile-verified, matrix operand order checked by reading, not run on hardware
//
#include <hip/hip_runtime.h>
#include <hip/hip_bf16.h>
#include <math.h>

// NADE forward as a chunked prefix-scan on CDNA5 WMMA pipes.
// B=512, D=784, H=512, C=2. 14 chunks x 56 steps (padded to 64 for GEMM).
//
// Pipeline:
//   prep_transpose : Wt[d][h] f32, Vt[d][c][h] bf16, xt[d][b] f32
//   prep_pad       : xpad[b][896] bf16, Wpad[h][896] bf16 (chunk-padded, zeros)
//   nade_chunk_gemm: S[k] = x_chunk @ W_chunkT via v_wmma_f32_16x16x32_bf16
//   nade_chunk_scan: exclusive scan over chunks, seeded with bias c
//   nade_main_scan : per-chunk sequential scan, WMMA logits + log_softmax

typedef __attribute__((ext_vector_type(16))) __bf16 v16bf;
typedef __attribute__((ext_vector_type(8)))  __bf16 v8bf;   // 16B = one b128
typedef __attribute__((ext_vector_type(8)))  float  v8f;
typedef __attribute__((ext_vector_type(4)))  float  v4f;    // 16B = one b128

constexpr int Bn  = 512;
constexpr int Dn  = 784;
constexpr int Hn  = 512;
constexpr int Cn  = 2;
constexpr int NCH = 14;          // chunks along D
constexpr int Ln  = 56;          // steps per chunk (NCH*Ln == Dn)
constexpr int DP  = NCH * 64;    // padded D = 896

// Workspace layout (all regions 256B aligned by construction)
constexpr size_t S_OFF  = 0;                         // f32 [NCH][Bn][Hn]
constexpr size_t S_SZ   = (size_t)NCH * Bn * Hn * 4; // 14,680,064
constexpr size_t XT_OFF = S_OFF + S_SZ;              // f32 [Dn][Bn]
constexpr size_t XT_SZ  = (size_t)Dn * Bn * 4;
constexpr size_t WT_OFF = XT_OFF + XT_SZ;            // f32 [Dn][Hn]
constexpr size_t WT_SZ  = (size_t)Dn * Hn * 4;
constexpr size_t VT_OFF = WT_OFF + WT_SZ;            // bf16 [Dn][Cn][Hn]
constexpr size_t VT_SZ  = (size_t)Dn * Cn * Hn * 2;
constexpr size_t XP_OFF = VT_OFF + VT_SZ;            // bf16 [Bn][DP]
constexpr size_t XP_SZ  = (size_t)Bn * DP * 2;
constexpr size_t WP_OFF = XP_OFF + XP_SZ;            // bf16 [Hn][DP]
constexpr size_t WP_SZ  = (size_t)Hn * DP * 2;
// total ~21.3 MB

// ---------------------------------------------------------------------------
// Prep 1: transposes. One thread per (d, h) with Hn == Bn == 512.
// ---------------------------------------------------------------------------
__global__ __launch_bounds__(256)
void prep_transpose(const float* __restrict__ x, const float* __restrict__ V,
                    const float* __restrict__ W, float* __restrict__ xt,
                    float* __restrict__ Wt, __bf16* __restrict__ Vt) {
    const int t = blockIdx.x * blockDim.x + threadIdx.x;
    if (t >= Dn * Hn) return;
    const int d = t / Hn;
    const int h = t % Hn;
    Wt[(size_t)d * Hn + h] = W[(size_t)h * Dn + d];
    xt[(size_t)d * Bn + h] = x[(size_t)h * Dn + d];  // h plays batch index
    Vt[((size_t)d * Cn + 0) * Hn + h] = (__bf16)V[((size_t)h * Dn + d) * Cn + 0];
    Vt[((size_t)d * Cn + 1) * Hn + h] = (__bf16)V[((size_t)h * Dn + d) * Cn + 1];
}

// ---------------------------------------------------------------------------
// Prep 2: chunk-padded bf16 copies of x and W (rows x 896, zero tail per chunk)
// ---------------------------------------------------------------------------
__global__ __launch_bounds__(256)
void prep_pad(const float* __restrict__ x, const float* __restrict__ W,
              __bf16* __restrict__ xpad, __bf16* __restrict__ Wpad) {
    const int t = blockIdx.x * blockDim.x + threadIdx.x;
    if (t >= Bn * DP) return;
    const int r  = t / DP;
    const int dp = t % DP;
    const int k  = dp >> 6;
    const int o  = dp & 63;
    const bool valid = (o < Ln);
    const int d  = k * Ln + o;
    xpad[t] = valid ? (__bf16)x[(size_t)r * Dn + d] : (__bf16)0.0f;
    Wpad[t] = valid ? (__bf16)W[(size_t)r * Dn + d] : (__bf16)0.0f;
}

// ---------------------------------------------------------------------------
// Kernel 1: S[k][b][h] = sum_{d in chunk k} x[b,d]*W[h,d].
// One wave per 16x16 tile; padded K=64 -> 2 uniform bf16 WMMA steps.
// Per-lane operand layout = two contiguous 8-elem K-runs -> b128 loads.
// ---------------------------------------------------------------------------
__global__ __launch_bounds__(32)
void nade_chunk_gemm(const __bf16* __restrict__ xpad,
                     const __bf16* __restrict__ Wpad, float* __restrict__ S) {
    const int k    = blockIdx.x;
    const int b0   = blockIdx.y * 16;
    const int h0   = blockIdx.z * 16;
    const int lane = threadIdx.x;
    const int m    = lane & 15;
    const int hi   = lane >> 4;

    const __bf16* xrow = xpad + (size_t)(b0 + m) * DP + k * 64;
    const __bf16* wrow = Wpad + (size_t)(h0 + m) * DP + k * 64;

    v8f acc = {};
#pragma unroll
    for (int ks = 0; ks < 2; ++ks) {
        v16bf av, bv;
#pragma unroll
        for (int g = 0; g < 2; ++g) {
            const int kk0 = ks * 32 + g * 16 + hi * 8;
            const v8bf ar = *(const v8bf*)(xrow + kk0);
            const v8bf br = *(const v8bf*)(wrow + kk0);
#pragma unroll
            for (int e = 0; e < 8; ++e) {
                av[8 * g + e] = ar[e];
                bv[8 * g + e] = br[e];
            }
        }
        acc = __builtin_amdgcn_wmma_f32_16x16x32_bf16(
            false, av, false, bv, (short)0, acc, false, false);
    }
#pragma unroll
    for (int r = 0; r < 8; ++r)
        S[((size_t)k * Bn + b0 + r + 8 * hi) * Hn + h0 + m] = acc[r];
}

// ---------------------------------------------------------------------------
// Kernel 2: in-place exclusive scan over chunks, seeded with bias c.
// ---------------------------------------------------------------------------
__global__ __launch_bounds__(256)
void nade_chunk_scan(float* __restrict__ S, const float* __restrict__ cbias) {
    const int t = blockIdx.x * blockDim.x + threadIdx.x;
    if (t >= Bn * Hn) return;
    const int b = t / Hn;
    const int h = t % Hn;
    float run = cbias[h];
#pragma unroll
    for (int k = 0; k < NCH; ++k) {
        const size_t idx = ((size_t)k * Bn + b) * Hn + h;
        const float tmp = S[idx];
        S[idx] = run;
        run += tmp;
    }
}

// ---------------------------------------------------------------------------
// Kernel 3: per-chunk sequential scan. Workgroup = 16 batch rows x 1 chunk;
// 8 waves own 64-wide H slices; 16x64 f32 a-state in VGPRs in WMMA A order.
// ---------------------------------------------------------------------------
__global__ __launch_bounds__(256)
void nade_main_scan(const float* __restrict__ xt, const __bf16* __restrict__ Vt,
                    const float* __restrict__ bbias, const float* __restrict__ Wt,
                    const float* __restrict__ Astart, float* __restrict__ out) {
    __shared__ float part[8 * 32];

    const int k    = blockIdx.x;
    const int b0   = blockIdx.y * 16;
    const int w    = threadIdx.x >> 5;
    const int lane = threadIdx.x & 31;
    const int m    = lane & 15;   // batch row (A) / logits col (B)
    const int hi   = lane >> 4;
    const int hb   = w * 64;      // this wave's H slice

    // Chunk-start state: 32 f32 per lane via contiguous 8-float runs.
    float a[2][16];
    {
        const float* arow = Astart + ((size_t)k * Bn + b0 + m) * Hn;
#pragma unroll
        for (int ks = 0; ks < 2; ++ks)
#pragma unroll
            for (int g = 0; g < 2; ++g) {
                const int hrun = hb + ks * 32 + g * 16 + hi * 8;
                const v4f a0 = *(const v4f*)(arow + hrun);
                const v4f a1 = *(const v4f*)(arow + hrun + 4);
#pragma unroll
                for (int e = 0; e < 4; ++e) {
                    a[ks][8 * g + e]     = a0[e];
                    a[ks][8 * g + 4 + e] = a1[e];
                }
            }
    }

    float* __restrict__ yout = out;
    float* __restrict__ pout = out + (size_t)Bn * Dn * Cn;

    for (int dd = 0; dd < Ln; ++dd) {
        const int d = k * Ln + dd;
        const float xv = xt[(size_t)d * Bn + b0 + m];     // coalesced
        const float* wd  = Wt + (size_t)d * Hn;
        const __bf16* vd = Vt + ((size_t)d * Cn + m) * Hn; // valid only if m<Cn

        v8f cacc = {};
#pragma unroll
        for (int ks = 0; ks < 2; ++ks) {
            v16bf av, bv;
#pragma unroll
            for (int g = 0; g < 2; ++g) {
                const int hrun = hb + ks * 32 + g * 16 + hi * 8;
                const v4f w0 = *(const v4f*)(wd + hrun);
                const v4f w1 = *(const v4f*)(wd + hrun + 4);
                // sigmoid -> bf16 A operand, then rank-1 update of a-state
#pragma unroll
                for (int e = 0; e < 4; ++e) {
                    const int e0 = 8 * g + e, e1 = 8 * g + 4 + e;
                    const float av0 = a[ks][e0], av1 = a[ks][e1];
                    av[e0] = (__bf16)(1.0f / (1.0f + __expf(-av0)));
                    av[e1] = (__bf16)(1.0f / (1.0f + __expf(-av1)));
                    a[ks][e0] = fmaf(xv, w0[e], av0);
                    a[ks][e1] = fmaf(xv, w1[e], av1);
                }
                if (m < Cn) {
                    const v8bf vr = *(const v8bf*)(vd + hrun);
#pragma unroll
                    for (int e = 0; e < 8; ++e) bv[8 * g + e] = vr[e];
                } else {
#pragma unroll
                    for (int e = 0; e < 8; ++e) bv[8 * g + e] = (__bf16)0.0f;
                }
            }
            cacc = __builtin_amdgcn_wmma_f32_16x16x32_bf16(
                false, av, false, bv, (short)0, cacc, false, false);
        }

        // Per-wave 16x2 partial logits -> LDS
        if (m < Cn) {
#pragma unroll
            for (int r = 0; r < 8; ++r)
                part[w * 32 + (r + 8 * hi) * Cn + m] = cacc[r];
        }
        __syncthreads();

        if (threadIdx.x < 16) {
            const int row = threadIdx.x;
            float l0 = bbias[(size_t)d * Cn + 0];
            float l1 = bbias[(size_t)d * Cn + 1];
#pragma unroll
            for (int ww = 0; ww < 8; ++ww) {
                l0 += part[ww * 32 + row * 2 + 0];
                l1 += part[ww * 32 + row * 2 + 1];
            }
            const size_t base = (size_t)(b0 + row) * (Dn * Cn) + (size_t)d * Cn;
            yout[base + 0] = l0;
            yout[base + 1] = l1;
            const float mx  = fmaxf(l0, l1);
            const float lse = mx + __logf(__expf(l0 - mx) + __expf(l1 - mx));
            pout[base + 0] = l0 - lse;
            pout[base + 1] = l1 - lse;
        }
        __syncthreads();
    }
}

// ---------------------------------------------------------------------------
extern "C" void kernel_launch(void* const* d_in, const int* in_sizes, int n_in,
                              void* d_out, int out_size, void* d_ws, size_t ws_size,
                              hipStream_t stream) {
    (void)in_sizes; (void)n_in; (void)out_size; (void)ws_size;
    const float* x  = (const float*)d_in[0];  // (B, D)
    const float* V  = (const float*)d_in[1];  // (H, D, C)
    const float* bb = (const float*)d_in[2];  // (D, C)
    const float* W  = (const float*)d_in[3];  // (H, D)
    const float* cb = (const float*)d_in[4];  // (1, H)
    float* out = (float*)d_out;

    char* ws = (char*)d_ws;
    float*  S    = (float*)(ws + S_OFF);
    float*  xt   = (float*)(ws + XT_OFF);
    float*  Wt   = (float*)(ws + WT_OFF);
    __bf16* Vt   = (__bf16*)(ws + VT_OFF);
    __bf16* xpad = (__bf16*)(ws + XP_OFF);
    __bf16* Wpad = (__bf16*)(ws + WP_OFF);

    prep_transpose<<<(Dn * Hn + 255) / 256, 256, 0, stream>>>(x, V, W, xt, Wt, Vt);
    prep_pad<<<(Bn * DP + 255) / 256, 256, 0, stream>>>(x, W, xpad, Wpad);

    dim3 g1(NCH, Bn / 16, Hn / 16);
    nade_chunk_gemm<<<g1, 32, 0, stream>>>(xpad, Wpad, S);

    nade_chunk_scan<<<(Bn * Hn + 255) / 256, 256, 0, stream>>>(S, cb);

    dim3 g3(NCH, Bn / 16);
    nade_main_scan<<<g3, 256, 0, stream>>>(xt, Vt, bb, Wt, S, out);
}
